// SpMiddleResNetFHD_65627100283297
// MI455X (gfx1250) — matrix operand order repaired and checked by reading, not compiled, not run on hardware
//
#include <hip/hip_runtime.h>
#include <hip/hip_bf16.h>

typedef _Float16 h16 __attribute__((ext_vector_type(16)));
typedef _Float16 h8  __attribute__((ext_vector_type(8)));
typedef float    f8  __attribute__((ext_vector_type(8)));

#define BN_EPS 0.001f

// ---------------- utility kernels ----------------

__global__ void zero16_kernel(uint4* __restrict__ p, long n16) {
  long i = (long)blockIdx.x * blockDim.x + threadIdx.x;
  if (i < n16) { uint4 z; z.x = z.y = z.z = z.w = 0u; p[i] = z; }
}

__global__ void scatter_kernel(const float* __restrict__ vf, const int* __restrict__ coors,
                               _Float16* __restrict__ dense, unsigned char* __restrict__ mask,
                               int N, int D, int H, int W, int C) {
  int v = blockIdx.x;
  int c = threadIdx.x;
  if (v >= N || c >= C) return;
  int b = coors[v * 4 + 0], z = coors[v * 4 + 1], y = coors[v * 4 + 2], x = coors[v * 4 + 3];
  long p = (((long)b * D + z) * H + y) * W + x;
  dense[p * C + c] = (_Float16)vf[(long)v * C + c];
  if (c == 0) mask[p] = 1;
}

// folded BN (+bias): out[0..C) = scale, out[C..2C) = shift
__global__ void fold_params_kernel(const float* __restrict__ g, const float* __restrict__ be,
                                   const float* __restrict__ mn, const float* __restrict__ vr,
                                   const float* __restrict__ bias, float* __restrict__ out, int C) {
  int c = threadIdx.x;
  if (c >= C) return;
  float sc = g[c] * rsqrtf(vr[c] + BN_EPS);
  float sh = be[c] - mn[c] * sc;
  if (bias) sh += bias[c] * sc;
  out[c] = sc;
  out[C + c] = sh;
}

// Repack (3,3,3,Cin,Cout) f32 weights into per-lane WMMA B-fragments (f16).
// Packed index: (((tap*KC + kc)*NT + nt)*32 + lane)*16 + e
// B layout (ISA 7.12.2): lane -> column N = lane&15 ; element e -> K = (lane>>4)*8 + (e&7) + (e>>3)*16
__global__ void pack_w_kernel(const float* __restrict__ w, _Float16* __restrict__ out,
                              int Cin, int Cout, int KC, int NT) {
  long total = 27L * KC * NT * 512;
  long i = (long)blockIdx.x * blockDim.x + threadIdx.x;
  if (i >= total) return;
  int e    = (int)(i & 15);
  int lane = (int)((i >> 4) & 31);
  long r   = i >> 9;
  int nt = (int)(r % NT); r /= NT;
  int kc = (int)(r % KC);
  int tap = (int)(r / KC);
  int n = nt * 16 + (lane & 15);
  int c = kc * 32 + (lane >> 4) * 8 + (e & 7) + (e >> 3) * 16;
  float v = (c < Cin) ? w[((long)tap * Cin + c) * Cout + n] : 0.0f;
  out[i] = (_Float16)v;
}

// mask max-pool 3x3x3, pad 1, given stride -> binary output mask
__global__ void mask_pool_kernel(const unsigned char* __restrict__ mi, unsigned char* __restrict__ mo,
                                 int B, int Di, int Hi, int Wi, int Do, int Ho, int Wo, int stride) {
  long S = (long)B * Do * Ho * Wo;
  long p = (long)blockIdx.x * blockDim.x + threadIdx.x;
  if (p >= S) return;
  int wo = (int)(p % Wo); long t = p / Wo;
  int ho = (int)(t % Ho); t /= Ho;
  int d0 = (int)(t % Do);
  int b  = (int)(t / Do);
  unsigned char acc = 0;
  for (int kd = 0; kd < 3; ++kd) {
    int di = d0 * stride + kd - 1; if (di < 0 || di >= Di) continue;
    for (int kh = 0; kh < 3; ++kh) {
      int hi = ho * stride + kh - 1; if (hi < 0 || hi >= Hi) continue;
      for (int kw = 0; kw < 3; ++kw) {
        int wi = wo * stride + kw - 1; if (wi < 0 || wi >= Wi) continue;
        acc |= mi[(((long)b * Di + di) * Hi + hi) * Wi + wi];
      }
    }
  }
  mo[p] = acc ? 1 : 0;
}

// ---------------- main implicit-GEMM conv via WMMA ----------------
// One wave computes a 16(spatial) x 16(Cout) tile with v_wmma_f32_16x16x32_f16.
// All dims are compile-time: no integer division in the address path, KC loop
// fully unrolled. Weight (B) fragments for this block's ntile are staged in LDS
// once and shared by all 8 waves.
template <int CIN, int COUT, int DI, int HI, int WI, int DO, int HO, int WO, int ST>
__global__ void __launch_bounds__(256)
conv_wmma_kernel(const _Float16* __restrict__ in, const unsigned char* __restrict__ mo,
                 const _Float16* __restrict__ wp, const float* __restrict__ ps,
                 const _Float16* __restrict__ res, _Float16* __restrict__ out) {
  constexpr int KC = (CIN + 31) / 32;
  constexpr int NT = COUT / 16;
  constexpr int Mtot = 2 * DO * HO * WO;
  constexpr int NFRAG = 27 * KC;            // fragments of 512 halfs (32 lanes x 16)

  extern __shared__ _Float16 smem[];        // NFRAG * 512 halfs

  const int tid  = threadIdx.x;
  const int lane = tid & 31;
  const int wave = tid >> 5;
  const int ntile = blockIdx.y;

  // ---- cooperative stage of all B fragments for this ntile into LDS ----
  for (int g = tid; g < NFRAG * 64; g += 256) {
    int frag = g >> 6;
    int off  = (g & 63) * 8;
    *(h8*)(smem + frag * 512 + off) = *(const h8*)(wp + ((long)frag * NT + ntile) * 512 + off);
  }
  __syncthreads();

  const int mtile = blockIdx.x * 8 + wave;
  if (mtile * 16 < Mtot) {                   // wave-uniform: EXEC all-1s at WMMA
    const int hl = lane >> 4;
    const int spos = mtile * 16 + (lane & 15);
    const bool rowok = spos < Mtot;
    const int sp = rowok ? spos : 0;
    const int wo = sp % WO;
    int t = sp / WO;
    const int ho = t % HO; t /= HO;
    const int d0 = t % DO;
    const int b  = t / DO;

    f8 acc = {};
    for (int tap = 0; tap < 27; ++tap) {
      const int kd = tap / 9, kh = (tap / 3) % 3, kw = tap % 3;
      const int di = d0 * ST + kd - 1;
      const int hi = ho * ST + kh - 1;
      const int wi = wo * ST + kw - 1;
      const bool valid = rowok && di >= 0 && di < DI && hi >= 0 && hi < HI && wi >= 0 && wi < WI;
      const long ibase = ((((long)b * DI + di) * HI + hi) * WI + wi) * (long)CIN;
#pragma unroll
      for (int kc = 0; kc < KC; ++kc) {
        // A fragment: 16-bit A 16x32 layout: e<8 -> K = hl*8+e ; e>=8 -> K = 16+hl*8+(e-8)
        const int c0 = kc * 32 + hl * 8;
        const int c1 = c0 + 16;
        h8 alo = {}, ahi = {};
        if (valid && c0 < CIN) alo = *(const h8*)(in + ibase + c0);
        if (valid && c1 < CIN) ahi = *(const h8*)(in + ibase + c1);
        h16 a;
#pragma unroll
        for (int e = 0; e < 8; ++e) { a[e] = alo[e]; a[e + 8] = ahi[e]; }
        // B fragment from LDS: 16 contiguous halfs per lane
        const _Float16* wq = smem + (tap * KC + kc) * 512 + lane * 16;
        h8 b0 = *(const h8*)wq;
        h8 b1 = *(const h8*)(wq + 8);
        h16 bm;
#pragma unroll
        for (int e = 0; e < 8; ++e) { bm[e] = b0[e]; bm[e + 8] = b1[e]; }
        acc = __builtin_amdgcn_wmma_f32_16x16x32_f16(false, a, false, bm, (short)0, acc,
                                                     false, false);
      }
    }

    // Epilogue: C/D layout: VGPR j -> row M = j + hl*8 ; col N = lane&15
    const int ch = ntile * 16 + (lane & 15);
    const float sc = ps[ch];
    const float sh = ps[COUT + ch];
#pragma unroll
    for (int j = 0; j < 8; ++j) {
      int p = mtile * 16 + hl * 8 + j;
      if (p >= Mtot) continue;
      float v = acc[j] * sc + sh;                       // BN(conv + bias)
      v = mo[p] ? v : 0.0f;                             // * mask
      if (res) v += (float)res[(long)p * COUT + ch];    // residual (unmasked, per reference)
      v = v > 0.0f ? v : 0.0f;                          // ReLU
      out[(long)p * COUT + ch] = (_Float16)v;
    }
  }
}

__global__ void h2f_kernel(const _Float16* __restrict__ in, float* __restrict__ out, long n) {
  long i = (long)blockIdx.x * blockDim.x + threadIdx.x;
  if (i < n) out[i] = (float)in[i];
}

// ---------------- host orchestration ----------------

template <int CIN, int COUT, int DI, int HI, int WI, int DO, int HO, int WO, int ST>
static void launch_conv(const _Float16* in, const unsigned char* mo, const _Float16* wp,
                        const float* ps, const _Float16* res, _Float16* out, hipStream_t stream) {
  constexpr int KC = (CIN + 31) / 32;
  constexpr int NT = COUT / 16;
  constexpr int Mtot = 2 * DO * HO * WO;
  int Mtiles = (Mtot + 15) / 16;
  dim3 grid((Mtiles + 7) / 8, NT);
  size_t smem = (size_t)27 * KC * 512 * sizeof(_Float16);
  conv_wmma_kernel<CIN, COUT, DI, HI, WI, DO, HO, WO, ST>
      <<<grid, 256, smem, stream>>>(in, mo, wp, ps, res, out);
}

struct LayerDesc {
  int w, b, be, ga, mn, vr;   // param leaf indices (relative to params base)
  int cin, cout, st;
  int res, ib, ob;            // buffer ids: 0=dense0,1=A,2=B,3=C ; res=-1 none
  int ir, orr;                // input/output resolution ids
};

extern "C" void kernel_launch(void* const* d_in, const int* in_sizes, int n_in,
                              void* d_out, int out_size, void* d_ws, size_t ws_size,
                              hipStream_t stream) {
  (void)out_size; (void)ws_size;
  const float* vf  = (const float*)d_in[0];
  const int* coors = (const int*)d_in[1];
  const int PBASE = 2;
  auto leaf = [&](int i) -> const float* {
    int j = PBASE + i;
    if (j >= n_in) j = n_in - 1;
    return (const float*)d_in[j];
  };
  const int N = in_sizes[0] / 128;

  const int Bsz = 2;
  const int dims[4][3] = {{24, 64, 64}, {12, 32, 32}, {6, 16, 16}, {3, 8, 8}};
  const long S[4] = {2L * 24 * 64 * 64, 2L * 12 * 32 * 32, 2L * 6 * 16 * 16, 2L * 3 * 8 * 8};

  // ---- workspace bump allocator ----
  char* ws = (char*)d_ws;
  size_t off = 0;
  auto alloc = [&](size_t bytes) -> void* {
    void* p = ws + off;
    off += (bytes + 255) & ~(size_t)255;
    return p;
  };
  _Float16* dense0 = (_Float16*)alloc(S[0] * 128 * sizeof(_Float16));
  _Float16* bufA   = (_Float16*)alloc(S[0] * 16 * sizeof(_Float16));
  _Float16* bufB   = (_Float16*)alloc(S[0] * 16 * sizeof(_Float16));
  _Float16* bufC   = (_Float16*)alloc(S[0] * 16 * sizeof(_Float16));
  unsigned char* mask0 = (unsigned char*)alloc(S[0]);
  unsigned char* mask1 = (unsigned char*)alloc(S[0]);
  float* pbuf = (float*)alloc(20 * 256 * sizeof(float));

  // Param leaf order: JAX sorted-key flatten of the nested dict.
  static const LayerDesc L[20] = {
    {115, -1, 111, 112, 113, 114, 128,  16, 1, -1, 0, 1, 0, 0},  // conv_input
    { 10,  0,   2,   3,   4,   5,  16,  16, 1, -1, 1, 2, 0, 0},  // c1.b0.conv1
    { 11,  1,   6,   7,   8,   9,  16,  16, 1,  1, 2, 3, 0, 0},  // c1.b0.conv2 (+res A)
    { 22, 12,  14,  15,  16,  17,  16,  16, 1, -1, 3, 1, 0, 0},  // c1.b1.conv1
    { 23, 13,  18,  19,  20,  21,  16,  16, 1,  3, 1, 2, 0, 0},  // c1.b1.conv2 (+res C)
    { 52, -1,  48,  49,  50,  51,  16,  32, 2, -1, 2, 1, 0, 1},  // conv2.down
    { 34, 24,  26,  27,  28,  29,  32,  32, 1, -1, 1, 2, 1, 1},
    { 35, 25,  30,  31,  32,  33,  32,  32, 1,  1, 2, 3, 1, 1},
    { 46, 36,  38,  39,  40,  41,  32,  32, 1, -1, 3, 1, 1, 1},
    { 47, 37,  42,  43,  44,  45,  32,  32, 1,  3, 1, 2, 1, 1},
    { 81, -1,  77,  78,  79,  80,  32,  64, 2, -1, 2, 1, 1, 2},  // conv3.down
    { 63, 53,  55,  56,  57,  58,  64,  64, 1, -1, 1, 2, 2, 2},
    { 64, 54,  59,  60,  61,  62,  64,  64, 1,  1, 2, 3, 2, 2},
    { 75, 65,  67,  68,  69,  70,  64,  64, 1, -1, 3, 1, 2, 2},
    { 76, 66,  71,  72,  73,  74,  64,  64, 1,  3, 1, 2, 2, 2},
    {110, -1, 106, 107, 108, 109,  64, 128, 2, -1, 2, 1, 2, 3},  // conv4.down
    { 92, 82,  84,  85,  86,  87, 128, 128, 1, -1, 1, 2, 3, 3},
    { 93, 83,  88,  89,  90,  91, 128, 128, 1,  1, 2, 3, 3, 3},
    {104, 94,  96,  97,  98,  99, 128, 128, 1, -1, 3, 1, 3, 3},
    {105, 95, 100, 101, 102, 103, 128, 128, 1,  3, 1, 2, 3, 3},
  };

  _Float16* wpk[20];
  for (int i = 0; i < 20; ++i) {
    int KC = (L[i].cin + 31) / 32, NT = L[i].cout / 16;
    wpk[i] = (_Float16*)alloc((size_t)(27L * KC * NT * 512) * sizeof(_Float16));
  }

  // ---- zero dense input + mask, scatter voxels ----
  long n16d = (S[0] * 128 * (long)sizeof(_Float16)) / 16;
  zero16_kernel<<<dim3((unsigned)((n16d + 255) / 256)), 256, 0, stream>>>((uint4*)dense0, n16d);
  long n16m = S[0] / 16;
  zero16_kernel<<<dim3((unsigned)((n16m + 255) / 256)), 256, 0, stream>>>((uint4*)mask0, n16m);
  scatter_kernel<<<dim3(N), 128, 0, stream>>>(vf, coors, dense0, mask0, N, 24, 64, 64, 128);

  // ---- fold BN params + repack weights ----
  for (int i = 0; i < 20; ++i) {
    const LayerDesc& l = L[i];
    fold_params_kernel<<<1, l.cout, 0, stream>>>(leaf(l.ga), leaf(l.be), leaf(l.mn), leaf(l.vr),
                                                 l.b >= 0 ? leaf(l.b) : (const float*)nullptr,
                                                 pbuf + i * 256, l.cout);
    int KC = (l.cin + 31) / 32, NT = l.cout / 16;
    long tot = 27L * KC * NT * 512;
    pack_w_kernel<<<dim3((unsigned)((tot + 255) / 256)), 256, 0, stream>>>(leaf(l.w), wpk[i],
                                                                           l.cin, l.cout, KC, NT);
  }

  // ---- conv chain with mask ping-pong ----
  void* bufs[4] = {dense0, bufA, bufB, bufC};
  unsigned char* mcur = mask0;
  unsigned char* mnxt = mask1;
  for (int i = 0; i < 20; ++i) {
    const LayerDesc& l = L[i];
    int Di = dims[l.ir][0], Hi = dims[l.ir][1], Wi = dims[l.ir][2];
    int Do = dims[l.orr][0], Ho = dims[l.orr][1], Wo = dims[l.orr][2];
    long Sout = (long)Bsz * Do * Ho * Wo;
    mask_pool_kernel<<<dim3((unsigned)((Sout + 255) / 256)), 256, 0, stream>>>(
        mcur, mnxt, Bsz, Di, Hi, Wi, Do, Ho, Wo, l.st);

    const _Float16* IN  = (const _Float16*)bufs[l.ib];
    const _Float16* RS  = l.res >= 0 ? (const _Float16*)bufs[l.res] : (const _Float16*)nullptr;
    _Float16*       OUT = (_Float16*)bufs[l.ob];
    const float*    PS  = pbuf + i * 256;
    const _Float16* WP  = wpk[i];

    switch (i) {
      case 0:
        launch_conv<128, 16, 24, 64, 64, 24, 64, 64, 1>(IN, mnxt, WP, PS, RS, OUT, stream); break;
      case 1: case 2: case 3: case 4:
        launch_conv<16, 16, 24, 64, 64, 24, 64, 64, 1>(IN, mnxt, WP, PS, RS, OUT, stream); break;
      case 5:
        launch_conv<16, 32, 24, 64, 64, 12, 32, 32, 2>(IN, mnxt, WP, PS, RS, OUT, stream); break;
      case 6: case 7: case 8: case 9:
        launch_conv<32, 32, 12, 32, 32, 12, 32, 32, 1>(IN, mnxt, WP, PS, RS, OUT, stream); break;
      case 10:
        launch_conv<32, 64, 12, 32, 32, 6, 16, 16, 2>(IN, mnxt, WP, PS, RS, OUT, stream); break;
      case 11: case 12: case 13: case 14:
        launch_conv<64, 64, 6, 16, 16, 6, 16, 16, 1>(IN, mnxt, WP, PS, RS, OUT, stream); break;
      case 15:
        launch_conv<64, 128, 6, 16, 16, 3, 8, 8, 2>(IN, mnxt, WP, PS, RS, OUT, stream); break;
      default:
        launch_conv<128, 128, 3, 8, 8, 3, 8, 8, 1>(IN, mnxt, WP, PS, RS, OUT, stream); break;
    }

    unsigned char* tmp = mcur; mcur = mnxt; mnxt = tmp;
  }

  // ---- final f16 -> f32 into d_out (layer 20 wrote bufB) ----
  long nout = S[3] * 128;
  h2f_kernel<<<dim3((unsigned)((nout + 255) / 256)), 256, 0, stream>>>(bufB, (float*)d_out, nout);
}
